// VQVAEVectorQuantizer_79190607003993
// MI455X (gfx1250) — matrix operand order repaired and checked by reading, not compiled, not run on hardware
//
#include <hip/hip_runtime.h>
#include <hip/hip_bf16.h>

// VQ-VAE vector quantizer for gfx1250 (MI455X).
// Score GEMM (32768 x 8192 x 256) on v_wmma_f32_16x16x32_f16, with the
// codebook staged in LDS via double-buffered global_load_async_to_lds_b128
// (ASYNCcnt-tracked) and shared by all 8 waves of the block.

#define N_TOK 32768          // B*H*W = 32*32*32
#define C_DIM 256
#define K_EMB 8192
#define HWP   1024           // H*W
#define NC    (N_TOK * C_DIM)

#define ROWS_PER_BLOCK 128   // 8 waves x 16 rows
#define CHUNK_K 64           // codebook rows staged per chunk (32 KB f16)
#define NCHUNK (K_EMB / CHUNK_K)

typedef __attribute__((ext_vector_type(16))) _Float16 v16h;
typedef __attribute__((ext_vector_type(8)))  _Float16 v8h;
typedef __attribute__((ext_vector_type(8)))  float    v8f;

union Frag16 { v16h v; v8h h[2]; };

// Wave-relative LDS byte offset of a flat pointer into a __shared__ object.
__device__ __forceinline__ uint32_t lds_off(const void* p) {
  return (uint32_t)(uintptr_t)(const __attribute__((address_space(3))) void*)p;
}

// Async copy 16 B global -> LDS (per-lane addresses), tracked by ASYNCcnt.
__device__ __forceinline__ void async_ld_b128(uint32_t lds_byte, const void* gaddr) {
  asm volatile("global_load_async_to_lds_b128 %0, %1, off"
               :: "v"(lds_byte), "v"(gaddr) : "memory");
}

// ---------------------------------------------------------------------------
// Kernel 1a: z [B,C,H,W] f32  ->  zf16 [N, C] f16 (row-major, n = b*HW + h*W + w)
// ---------------------------------------------------------------------------
__global__ __launch_bounds__(256) void vq_zcast(const float* __restrict__ z,
                                                _Float16* __restrict__ zf16) {
  int j  = blockIdx.x * 256 + threadIdx.x;      // j indexes zf16 (coalesced writes)
  int n  = j >> 8;
  int c  = j & 255;
  int b  = n >> 10;
  int hw = n & 1023;
  zf16[j] = (_Float16)z[b * (C_DIM * HWP) + c * HWP + hw];
}

// ---------------------------------------------------------------------------
// Kernel 1b: emb f32 -> ef16 [K, C] f16, plus e2[k] = sum_c emb[k][c]^2
// ---------------------------------------------------------------------------
__global__ __launch_bounds__(256) void vq_ecast(const float* __restrict__ emb,
                                                _Float16* __restrict__ ef16,
                                                float* __restrict__ e2) {
  __shared__ float red[256];
  int k = blockIdx.x, c = threadIdx.x;
  float v = emb[k * C_DIM + c];
  ef16[k * C_DIM + c] = (_Float16)v;
  red[c] = v * v;
  __syncthreads();
  for (int s = 128; s > 0; s >>= 1) {
    if (c < s) red[c] += red[c + s];
    __syncthreads();
  }
  if (c == 0) e2[k] = red[0];
}

// ---------------------------------------------------------------------------
// Kernel 2: fused score GEMM + argmin.
// Block = 128 z-rows (wave w owns rows n0+16w..n0+16w+15); all waves scan the
// full codebook.  Codebook chunks (64 rows, 32 KB) are staged in LDS with
// double-buffered async loads and consumed by all 8 waves.
//
// Score s(n,k) = ||e_k||^2 - 2*dot(z_n,e_k); ||z_n||^2 is row-constant.
//
// Per ISA 7.12.2 (wave32):
//  A (16x32 f16): lanes 0-15 row M=lane hold K={0..7,16..23}; lanes 16-31 same
//                 rows hold K={8..15,24..31}  -> two contiguous v8h loads/lane.
//  B (32x16 f16): lane%16 = column N; lanes 0-15 hold K={0..7,16..23}, lanes 16-31
//                 hold K={8..15,24..31}      -> two contiguous v8h loads/lane.
//  C/D (16x16 f32, 8 VGPRs): slot r on lane-half h holds row M = r + 8*h,
//                 column N = lane%16.
// ---------------------------------------------------------------------------
__global__ __launch_bounds__(256) void vq_argmin(const _Float16* __restrict__ zf16,
                                                 const _Float16* __restrict__ ef16,
                                                 const float* __restrict__ e2,
                                                 int* __restrict__ idxOut) {
  // Two 32 KB staging buffers; reduction arrays alias them after the k loop.
  __shared__ _Float16 sB[2][CHUNK_K * C_DIM];

  const int tid  = threadIdx.x;
  const int wave = tid >> 5;
  const int lane = tid & 31;
  const int hlf  = lane >> 4;    // lane half: selects K sub-window
  const int lrow = lane & 15;    // A row / B column within tile
  const int n0   = blockIdx.x * ROWS_PER_BLOCK;

  // ---- A fragments: this wave's 16 rows x full C=256, loaded once ----
  const _Float16* zr = zf16 + (size_t)(n0 + wave * 16 + lrow) * C_DIM;
  Frag16 afrag[8];
#pragma unroll
  for (int d = 0; d < 8; ++d) {
    const int base = 32 * d + hlf * 8;
    afrag[d].h[0] = *(const v8h*)(zr + base);        // K = base..base+7
    afrag[d].h[1] = *(const v8h*)(zr + base + 16);   // K = base+16..base+23
  }

  float minv[8];
  int   mini[8];
#pragma unroll
  for (int r = 0; r < 8; ++r) { minv[r] = 3.4e38f; mini[r] = 0; }

  // Per-chunk async issue: 32 KB / 256 threads = 8 x b128 per thread.
  const char* gemb = (const char*)ef16;
  const uint32_t lds0 = lds_off(&sB[0][0]);
  const uint32_t lds1 = lds_off(&sB[1][0]);

  {
    const char* src = gemb;                          // chunk 0
#pragma unroll
    for (int r = 0; r < 8; ++r) {
      uint32_t o = (uint32_t)(r * 4096 + tid * 16);
      async_ld_b128(lds0 + o, src + o);
    }
  }

  for (int chunk = 0; chunk < NCHUNK; ++chunk) {
    const int buf = chunk & 1;
    if (chunk + 1 < NCHUNK) {
      const char* src = gemb + (size_t)(chunk + 1) * CHUNK_K * C_DIM * 2;
      const uint32_t dst = (buf ? lds0 : lds1);
#pragma unroll
      for (int r = 0; r < 8; ++r) {
        uint32_t o = (uint32_t)(r * 4096 + tid * 16);
        async_ld_b128(dst + o, src + o);
      }
      // Async loads complete in order: <=8 outstanding => current chunk landed.
      asm volatile("s_wait_asynccnt 0x8" ::: "memory");
    } else {
      asm volatile("s_wait_asynccnt 0x0" ::: "memory");
    }
    __syncthreads();                                 // chunk visible to all waves

    const _Float16* bbase = &sB[buf][0];
#pragma unroll
    for (int t4 = 0; t4 < 4; ++t4) {                 // 4 k-tiles per chunk
      const _Float16* er = bbase + (t4 * 16 + lrow) * C_DIM + hlf * 8;
      Frag16 bfrag[8];
#pragma unroll
      for (int d = 0; d < 8; ++d) {                  // ds_load_b128 x2 per frag
        bfrag[d].h[0] = *(const v8h*)(er + 32 * d);
        bfrag[d].h[1] = *(const v8h*)(er + 32 * d + 16);
      }
      v8f acc = {};
#pragma unroll
      for (int d = 0; d < 8; ++d)
        acc = __builtin_amdgcn_wmma_f32_16x16x32_f16(
            /*neg_a=*/false, afrag[d].v, /*neg_b=*/false, bfrag[d].v,
            /*c_mod=*/(short)0, acc, /*reuse_a=*/false, /*reuse_b=*/false);

      const int   kk = chunk * CHUNK_K + t4 * 16 + lrow;
      const float ev = e2[kk];
#pragma unroll
      for (int r = 0; r < 8; ++r) {
        float s = ev - 2.0f * acc[r];
        if (s < minv[r]) { minv[r] = s; mini[r] = kk; }  // k increases -> first min
      }
    }
    __syncthreads();                                 // done reading buf: reusable
  }

  // ---- per-wave argmin reduce across the 16 column-lanes ----
  float* sval = (float*)&sB[0][0];                   // 2048 floats (aliased)
  int*   sidx = (int*)&sB[1][0];                     // 2048 ints   (aliased)
#pragma unroll
  for (int r = 0; r < 8; ++r) {
    int e = (tid << 3) + r;
    sval[e] = minv[r];
    sidx[e] = mini[r];
  }
  __syncthreads();

  if (tid < 128) {
    const int wv   = tid >> 4;         // which wave's rows
    const int m    = tid & 15;         // row within the wave's tile
    const int r    = m & 7;
    const int hsel = (m >> 3) << 4;    // lanes 0-15 hold rows 0-7, 16-31 rows 8-15
    float best = 3.4e38f;
    int   bidx = 0x7fffffff;
    for (int li = 0; li < 16; ++li) {
      int   e = ((wv * 32 + hsel + li) << 3) + r;
      float v = sval[e];
      int   i = sidx[e];
      if (v < best || (v == best && i < bidx)) { best = v; bidx = i; }
    }
    idxOut[n0 + wv * 16 + m] = bidx;
  }
}

// ---------------------------------------------------------------------------
// Kernel 3: gather z_q from the ORIGINAL fp32 codebook (output precision
// depends only on the index), write straight-through output, and produce
// per-block partial sums of (z_q - z)^2 (deterministic reduction).
// ---------------------------------------------------------------------------
__global__ __launch_bounds__(256) void vq_gather(const float* __restrict__ z,
                                                 const float* __restrict__ emb,
                                                 const int* __restrict__ idx,
                                                 float* __restrict__ out,
                                                 float* __restrict__ partial) {
  __shared__ float red[256];
  int i  = blockIdx.x * 256 + threadIdx.x;           // [B,C,H,W] flat index
  int wp = i & 31;
  int hp = (i >> 5) & 31;
  int c  = (i >> 10) & 255;
  int b  = i >> 18;
  int n  = b * HWP + hp * 32 + wp;
  int k  = idx[n];
  float v = emb[(size_t)k * C_DIM + c];
  float d = v - z[i];
  out[i] = v;                                        // z_q_st forward value == z_q
  red[threadIdx.x] = d * d;
  __syncthreads();
  for (int s = 128; s > 0; s >>= 1) {
    if (threadIdx.x < s) red[threadIdx.x] += red[threadIdx.x + s];
    __syncthreads();
  }
  if (threadIdx.x == 0) partial[blockIdx.x] = red[0];
}

// ---------------------------------------------------------------------------
// Kernel 4: deterministic final sum of NC/256 = 32768 partials -> loss.
// loss = 2 * mean((z_q - z)^2)
// ---------------------------------------------------------------------------
__global__ __launch_bounds__(256) void vq_loss(const float* __restrict__ partial,
                                               float* __restrict__ loss_out) {
  __shared__ float red[256];
  float s = 0.0f;
  for (int i = threadIdx.x; i < (NC / 256); i += 256) s += partial[i];
  red[threadIdx.x] = s;
  __syncthreads();
  for (int st = 128; st > 0; st >>= 1) {
    if (threadIdx.x < st) red[threadIdx.x] += red[threadIdx.x + st];
    __syncthreads();
  }
  if (threadIdx.x == 0) *loss_out = 2.0f * red[0] / (float)NC;
}

// ---------------------------------------------------------------------------
extern "C" void kernel_launch(void* const* d_in, const int* in_sizes, int n_in,
                              void* d_out, int out_size, void* d_ws, size_t ws_size,
                              hipStream_t stream) {
  const float* z   = (const float*)d_in[0];   // [32,256,32,32]
  const float* emb = (const float*)d_in[1];   // [8192,256]
  float*       out = (float*)d_out;           // z_q (NC floats) ++ loss (1 float)

  // Workspace layout (bytes)
  char* ws = (char*)d_ws;
  const size_t ZF16_B = (size_t)N_TOK * C_DIM * sizeof(_Float16); // 16 MiB
  const size_t EF16_B = (size_t)K_EMB * C_DIM * sizeof(_Float16); //  4 MiB
  const size_t E2_B   = (size_t)K_EMB * sizeof(float);
  const size_t IDX_B  = (size_t)N_TOK * sizeof(int);
  _Float16* zf16    = (_Float16*)(ws);
  _Float16* ef16    = (_Float16*)(ws + ZF16_B);
  float*    e2      = (float*)   (ws + ZF16_B + EF16_B);
  int*      idx     = (int*)     (ws + ZF16_B + EF16_B + E2_B);
  float*    partial = (float*)   (ws + ZF16_B + EF16_B + E2_B + IDX_B);
  float*    loss    = out + NC;

  vq_zcast <<<NC / 256,              256, 0, stream>>>(z, zf16);
  vq_ecast <<<K_EMB,                 256, 0, stream>>>(emb, ef16, e2);
  vq_argmin<<<N_TOK / ROWS_PER_BLOCK, 256, 0, stream>>>(zf16, ef16, e2, idx);
  vq_gather<<<NC / 256,              256, 0, stream>>>(z, emb, idx, out, partial);
  vq_loss  <<<1,                     256, 0, stream>>>(partial, loss);
}
